// SparseKSelfAttention_42631845380804
// MI455X (gfx1250) — matrix operand order repaired and checked by reading, not compiled, not run on hardware
//
#include <hip/hip_runtime.h>
#include <hip/hip_bf16.h>

// ---------------- problem constants ----------------
#define TT   2048
#define DD   1024
#define HH   16
#define HD   64
#define NKV  4
#define KSEL 64
#define SELCAP 96

typedef __attribute__((ext_vector_type(16))) __bf16 v16bf;
typedef __attribute__((ext_vector_type(8)))  float  v8f;
typedef __attribute__((ext_vector_type(4)))  unsigned v4u;
typedef __attribute__((ext_vector_type(4)))  int      v4i;
typedef __attribute__((ext_vector_type(8)))  int      v8i;

#if __has_builtin(__builtin_amdgcn_tensor_load_to_lds) && \
    __has_builtin(__builtin_amdgcn_s_wait_tensorcnt)
#define HAVE_TDM 1
#else
#define HAVE_TDM 0
#endif

// f32 -> bf16 with round-to-nearest-even (pure bit manipulation)
static __device__ __forceinline__ __bf16 f2bf(float f) {
  unsigned u = __builtin_bit_cast(unsigned, f);
  unsigned r = (u + 0x7FFFu + ((u >> 16) & 1u)) >> 16;
  return __builtin_bit_cast(__bf16, (unsigned short)r);
}

// monotonic order-preserving f32 <-> u32 transform
static __device__ __forceinline__ unsigned f2ord(float f) {
  unsigned u = __builtin_bit_cast(unsigned, f);
  return (u & 0x80000000u) ? ~u : (u | 0x80000000u);
}
static __device__ __forceinline__ float ord2f(unsigned ou) {
  unsigned u = (ou & 0x80000000u) ? (ou & 0x7FFFFFFFu) : ~ou;
  return __builtin_bit_cast(float, u);
}

#if HAVE_TDM
// ---------------------------------------------------------------------------
// TDM: DMA a 2-D f32 tile (tile_d1 rows x tile_d0 cols, row stride `stride0`
// elements) from global memory into LDS at byte offset `lds_addr`, row-major
// packed.  Descriptor packing per CDNA5 ISA.
// ---------------------------------------------------------------------------
static __device__ __forceinline__ void tdm_load_2d(unsigned lds_addr,
                                                   const float* gptr,
                                                   unsigned tensor_d0,
                                                   unsigned tensor_d1,
                                                   unsigned tile_d0,
                                                   unsigned tile_d1,
                                                   unsigned long long stride0) {
  unsigned long long ga = (unsigned long long)(uintptr_t)gptr;
  v4u g0;
  g0[0] = 1u;                                            // count=1, user desc
  g0[1] = lds_addr;                                      // LDS byte address
  g0[2] = (unsigned)(ga & 0xFFFFFFFFu);                  // global addr [31:0]
  g0[3] = (unsigned)((ga >> 32) & 0x01FFFFFFu) | (2u << 30); // [56:32] | type=2
  v8i g1;
  g1[0] = (int)(2u << 16);                               // data_size = 4 bytes
  g1[1] = (int)((tensor_d0 & 0xFFFFu) << 16);            // tensor_dim0[15:0]
  g1[2] = (int)((tensor_d0 >> 16) | ((tensor_d1 & 0xFFFFu) << 16));
  g1[3] = (int)((tensor_d1 >> 16) | (tile_d0 << 16));    // | tile_dim0
  g1[4] = (int)tile_d1;                                  // tile_dim1 (dim2=0)
  g1[5] = (int)(unsigned)(stride0 & 0xFFFFFFFFull);      // dim0_stride[31:0]
  g1[6] = (int)(unsigned)((stride0 >> 32) & 0xFFFFull);  // dim0_stride[47:32]
  g1[7] = 0;                                             // dim1_stride unused
  v4i z4 = {0, 0, 0, 0};
#if defined(__clang_major__) && (__clang_major__ >= 23)
  v8i z8 = {0, 0, 0, 0, 0, 0, 0, 0};
  __builtin_amdgcn_tensor_load_to_lds(g0, g1, z4, z4, z8, 0);
#else
  __builtin_amdgcn_tensor_load_to_lds(g0, g1, z4, z4, 0);
#endif
}
#endif  // HAVE_TDM

// ============================================================================
// GEMM: C[M x N] = A[M x K] @ W[K x N] + bias[N]    (bf16 WMMA, f32 accum)
// block = 256 threads = 8 waves; each wave owns a 16x16 tile; block tile 16x128
// grid = (N/128, M/16).  Each wave double-buffers its A(16x32) and B(32x16)
// chunks in private LDS via TDM; s_wait_tensorcnt 2 overlaps DMA of chunk k+1
// with WMMA of chunk k.  No cross-wave barriers.
// ============================================================================
__global__ void gemm_bias_kernel(const float* __restrict__ A,
                                 const float* __restrict__ W,
                                 const float* __restrict__ bias,
                                 float* __restrict__ C,
                                 int M, int N, int K) {
  const int tid  = threadIdx.x;
  const int lane = tid & 31;
  const int wv   = tid >> 5;
  const int m0   = blockIdx.y * 16;
  const int n0   = blockIdx.x * 128 + wv * 16;
  const int r    = lane & 15;     // row lane (A) / col lane (B, C)
  const int hf   = lane >> 4;     // half-wave selector
  const int col  = n0 + r;        // C column for this lane

  v8f acc = {};

#if HAVE_TDM
  // per wave, per buffer: A tile 16x32 (512 f32) then B tile 32x16 (512 f32)
  __shared__ float stage[8][2][1024];

  // issue the TDM pair for K-chunk `kc` into buffer `b`
  auto issue = [&](int kc, int b) {
    unsigned lbase = (unsigned)(uintptr_t)&stage[wv][b][0];
    tdm_load_2d(lbase, A + (size_t)m0 * K + kc,
                (unsigned)K, (unsigned)M, 32u, 16u, (unsigned long long)K);
    tdm_load_2d(lbase + 512u * 4u, W + (size_t)kc * N + n0,
                (unsigned)N, (unsigned)K, 16u, 32u, (unsigned long long)N);
  };

  issue(0, 0);
  int buf = 0;
  for (int kc = 0; kc < K; kc += 32) {
    const bool more = (kc + 32) < K;
    if (more) issue(kc + 32, buf ^ 1);          // prefetch next chunk
    if (more) __builtin_amdgcn_s_wait_tensorcnt(2);  // current pair landed
    else      __builtin_amdgcn_s_wait_tensorcnt(0);
    // compiler fence: TDM writes LDS behind the compiler's back; force the
    // staged-fragment loads below to stay inside the loop, after the wait.
    __asm__ __volatile__("" ::: "memory");

    const float* ast = &stage[wv][buf][0];
    const float* bst = &stage[wv][buf][512];
    v16bf a, b;
    // A (16x32 bf16): lanes 0-15 -> K 0..15 ; lanes 16-31 -> K 16..31
#pragma unroll
    for (int e = 0; e < 16; ++e) {
      int kk = hf * 16 + e;
      a[e] = f2bf(ast[r * 32 + kk]);
    }
    // B (32x16 bf16): lanes 0-15 -> K 0..15 ; lanes 16-31 -> K 16..31
#pragma unroll
    for (int e = 0; e < 16; ++e) {
      b[e] = f2bf(bst[(hf * 16 + e) * 16 + r]);
    }
    acc = __builtin_amdgcn_wmma_f32_16x16x32_bf16(false, a, false, b,
                                                  (short)0, acc, false, false);
    buf ^= 1;
  }
#else
  // fallback: direct global fragment loads
  for (int kc = 0; kc < K; kc += 32) {
    if (kc + 32 < K)
      __builtin_prefetch(&W[(size_t)(kc + 32) * N + col], 0, 0);
    v16bf a, b;
#pragma unroll
    for (int e = 0; e < 16; ++e) {
      int kk = hf * 16 + e;
      a[e] = f2bf(A[(size_t)(m0 + r) * K + kc + kk]);
    }
#pragma unroll
    for (int e = 0; e < 16; ++e) {
      b[e] = f2bf(W[(size_t)(kc + hf * 16 + e) * N + col]);
    }
    acc = __builtin_amdgcn_wmma_f32_16x16x32_bf16(false, a, false, b,
                                                  (short)0, acc, false, false);
  }
#endif

  const float bv = bias[col];
  // C/D: VGPR i -> row i (lanes 0-15) / row 8+i (lanes 16-31)
#pragma unroll
  for (int i = 0; i < 8; ++i)
    C[(size_t)(m0 + hf * 8 + i) * N + col] = acc[i] + bv;
}

// ============================================================================
// per-(t,head) RMSNorm over HD=64, in place.  one wave32 per block.
// ============================================================================
__global__ void rmsnorm64_kernel(float* __restrict__ base,
                                 const float* __restrict__ w,
                                 int stride_t, int nh) {
  const int b = blockIdx.x;
  const int t = b / nh, h = b % nh;
  float* p = base + (size_t)t * stride_t + h * HD;
  const int lane = threadIdx.x;
  float x0 = p[lane], x1 = p[lane + 32];
  float ss = x0 * x0 + x1 * x1;
#pragma unroll
  for (int off = 16; off > 0; off >>= 1) ss += __shfl_xor(ss, off, 32);
  float rr = rsqrtf(ss * (1.0f / 64.0f) + 1e-8f);
  p[lane]      = x0 * rr * w[lane];
  p[lane + 32] = x1 * rr * w[lane + 32];
}

// ============================================================================
// score tile: raw 16x16 Q·K^T accumulator via two bf16 WMMAs (K = 64 = 2x32)
// ============================================================================
static __device__ __forceinline__ v8f score_tile(v16bf a0, v16bf a1,
                                                 const float* __restrict__ kvb,
                                                 int j, int kvh, int hf) {
  const float* kp = kvb + (size_t)j * (2 * NKV * HD) + kvh * HD;
  v16bf b0, b1;
#pragma unroll
  for (int e = 0; e < 16; ++e) {
    int d = hf * 16 + e;
    b0[e] = f2bf(kp[d]);
    b1[e] = f2bf(kp[32 + d]);
  }
  v8f acc = {};
  acc = __builtin_amdgcn_wmma_f32_16x16x32_bf16(false, a0, false, b0,
                                                (short)0, acc, false, false);
  acc = __builtin_amdgcn_wmma_f32_16x16x32_bf16(false, a1, false, b1,
                                                (short)0, acc, false, false);
  return acc;
}

// ============================================================================
// sparse top-64 attention.  block = 128 threads (4 waves) handles one head and
// one 16-query tile.  grid = (T/16, H).
// exact 64th-largest threshold per row via 4-pass radix select on ordered bits;
// scores recomputed per pass with WMMA (cheaper than materializing 16x2048 f32).
// ============================================================================
__global__ void attn_kernel(const float* __restrict__ qb,
                            const float* __restrict__ kvb,
                            float* __restrict__ yb) {
  __shared__ float    qs[16][HD];
  __shared__ unsigned hist[16][256];
  __shared__ unsigned rowmax_u[16];
  __shared__ unsigned prefix[16];
  __shared__ int      needed[16];
  __shared__ int      cnt[16];
  __shared__ float    rowsum[16];
  __shared__ unsigned sel_idx[16][SELCAP];
  __shared__ float    sel_p[16][SELCAP];

  const int tid  = threadIdx.x;
  const int lane = tid & 31;
  const int wv   = tid >> 5;
  const int bx   = blockIdx.x;       // query tile
  const int hh   = blockIdx.y;       // head
  const int q0   = bx * 16;
  const int kvh  = hh >> 2;          // GQA: head -> kv head (rep = 4)
  const int r    = lane & 15;
  const int hf   = lane >> 4;
  const int ntile = bx + 1;          // key tiles 0..bx touch the causal region
  const float slope = exp2f(-8.0f * (float)hh / 15.0f);

  // load 16x64 normed-q tile into LDS
  for (int i = tid; i < 16 * HD; i += 128) {
    int rr = i / HD, d = i % HD;
    qs[rr][d] = qb[(size_t)(q0 + rr) * DD + hh * HD + d];
  }
  if (tid < 16) {
    rowmax_u[tid] = 0u; prefix[tid] = 0u; needed[tid] = KSEL;
    cnt[tid] = 0; rowsum[tid] = 0.0f;
  }
  __syncthreads();

  // A fragments (reused for every key tile and every pass)
  v16bf a0, a1;
#pragma unroll
  for (int e = 0; e < 16; ++e) {
    int kk = hf * 16 + e;
    a0[e] = f2bf(qs[r][kk]);
    a1[e] = f2bf(qs[r][32 + kk]);
  }

  // -------- radix select: find per-row 64th-largest score (ordered bits) -----
  for (int pass = 3; pass >= 0; --pass) {
    for (int i = tid; i < 16 * 256; i += 128) ((unsigned*)hist)[i] = 0u;
    __syncthreads();

    const unsigned hmask = (pass == 3) ? 0u : (0xFFFFFFFFu << (8 * (pass + 1)));
    for (int jt = wv; jt < ntile; jt += 4) {
      const int j = jt * 16 + r;
      if (jt + 4 < ntile)
        __builtin_prefetch(kvb + (size_t)(jt * 16 + 64 + r) * (2 * NKV * HD) + kvh * HD, 0, 0);
      v8f acc = score_tile(a0, a1, kvb, j, kvh, hf);
#pragma unroll
      for (int i = 0; i < 8; ++i) {
        const int row = hf * 8 + i;
        const int qr  = q0 + row;
        float sc = acc[i] * 0.125f - slope * (float)(qr - j);
        const bool valid = (j <= qr);
        if (!valid) sc = -__builtin_inff();
        const unsigned ou = f2ord(sc);
        if (pass == 3 && valid) atomicMax(&rowmax_u[row], ou);
        if ((ou & hmask) == (prefix[row] & hmask))
          atomicAdd(&hist[row][(ou >> (8 * pass)) & 255u], 1u);
      }
    }
    __syncthreads();

    if (tid < 16) {                    // per-row descending digit scan
      const int row = tid;
      unsigned cum = 0;
      const unsigned nd = (unsigned)needed[row];
      for (int d = 255; d >= 0; --d) {
        const unsigned c = hist[row][d];
        cum += c;
        if (cum >= nd) {
          prefix[row] |= ((unsigned)d) << (8 * pass);
          needed[row] = (int)(nd - (cum - c));
          break;
        }
      }
    }
    __syncthreads();
  }

  // -------- final pass: gather selected keys, exp weights -------------------
  for (int jt = wv; jt < ntile; jt += 4) {
    const int j = jt * 16 + r;
    v8f acc = score_tile(a0, a1, kvb, j, kvh, hf);
#pragma unroll
    for (int i = 0; i < 8; ++i) {
      const int row = hf * 8 + i;
      const int qr  = q0 + row;
      if (j <= qr) {
        const float sc = acc[i] * 0.125f - slope * (float)(qr - j);
        const unsigned ou = f2ord(sc);
        if (ou >= prefix[row]) {
          const float mx = ord2f(rowmax_u[row]);
          const float pv = expf(sc - mx);
          const int slot = atomicAdd(&cnt[row], 1);
          if (slot < SELCAP) {
            sel_idx[row][slot] = (unsigned)j;
            sel_p[row][slot]   = pv;
            atomicAdd(&rowsum[row], pv);
          }
        }
      }
    }
  }
  __syncthreads();

  // -------- sparse A·V: 4 rows per wave, lane covers 2 dims -----------------
  const int d0 = lane * 2;
#pragma unroll
  for (int rr = 0; rr < 4; ++rr) {
    const int row = wv * 4 + rr;
    const int qr  = q0 + row;
    const int n   = min(cnt[row], SELCAP);
    const float inv = 1.0f / rowsum[row];
    float acc0 = 0.0f, acc1 = 0.0f;
    for (int s = 0; s < n; ++s) {
      const int   j  = (int)sel_idx[row][s];
      const float pw = sel_p[row][s] * inv;
      const float* vp = kvb + (size_t)j * (2 * NKV * HD) + NKV * HD + kvh * HD + d0;
      acc0 += pw * vp[0];
      acc1 += pw * vp[1];
    }
    yb[(size_t)qr * DD + hh * HD + d0]     = acc0;
    yb[(size_t)qr * DD + hh * HD + d0 + 1] = acc1;
  }
}

// ============================================================================
// launcher
// ============================================================================
extern "C" void kernel_launch(void* const* d_in, const int* in_sizes, int n_in,
                              void* d_out, int out_size, void* d_ws, size_t ws_size,
                              hipStream_t stream) {
  (void)in_sizes; (void)n_in; (void)out_size; (void)ws_size;
  const float* x    = (const float*)d_in[0];
  const float* wq   = (const float*)d_in[1];
  const float* bq   = (const float*)d_in[2];
  const float* wkv  = (const float*)d_in[3];
  const float* bkv  = (const float*)d_in[4];
  const float* wo   = (const float*)d_in[5];
  const float* bo   = (const float*)d_in[6];
  const float* qn_w = (const float*)d_in[7];
  const float* kn_w = (const float*)d_in[8];
  float* out = (float*)d_out;

  float* qbuf  = (float*)d_ws;                        // T x (H*HD)      = 2048x1024
  float* kvbuf = qbuf + (size_t)TT * DD;              // T x (2*NKV*HD)  = 2048x512
  float* ybuf  = kvbuf + (size_t)TT * (2 * NKV * HD); // T x (H*HD)      = 2048x1024

  // Q = x @ wq + bq
  gemm_bias_kernel<<<dim3(DD / 128, TT / 16), 256, 0, stream>>>(
      x, wq, bq, qbuf, TT, DD, DD);
  // KV = x @ wkv + bkv   (cols: [k(4x64) | v(4x64)])
  gemm_bias_kernel<<<dim3((2 * NKV * HD) / 128, TT / 16), 256, 0, stream>>>(
      x, wkv, bkv, kvbuf, TT, 2 * NKV * HD, DD);
  // rmsnorm q (per t,head) and k (per t,kv-head), in place
  rmsnorm64_kernel<<<TT * HH, 32, 0, stream>>>(qbuf, qn_w, DD, HH);
  rmsnorm64_kernel<<<TT * NKV, 32, 0, stream>>>(kvbuf, kn_w, 2 * NKV * HD, NKV);
  // sparse top-64 attention -> y (T x D, heads concatenated)
  attn_kernel<<<dim3(TT / 16, HH), 128, 0, stream>>>(qbuf, kvbuf, ybuf);
  // out = y @ wo + bo
  gemm_bias_kernel<<<dim3(DD / 128, TT / 16), 256, 0, stream>>>(
      ybuf, wo, bo, out, TT, DD, DD);
}
